// FedUnlMlp_31679678776006
// MI455X (gfx1250) — compile-verified, matrix operand order checked by reading, not run on hardware
//
#include <hip/hip_runtime.h>
#include <hip/hip_bf16.h>
#include <math.h>

// Problem constants (match reference)
#define B_SZ   1024
#define SEQ_L  50
#define H_DIM  512
#define H4     2048
#define NITEM  50000

typedef __attribute__((ext_vector_type(16))) __bf16       v16bf;
typedef __attribute__((ext_vector_type(8)))  float        v8f;
typedef __attribute__((ext_vector_type(4)))  unsigned int u32x4;
typedef __attribute__((ext_vector_type(8)))  unsigned int u32x8;

struct alignas(32) BfPair { u32x4 a, b; };

__device__ __forceinline__ v16bf make_frag(const unsigned short* p0,
                                           const unsigned short* p1) {
  BfPair p;
  p.a = *(const u32x4*)p0;   // 8 bf16 (16B)
  p.b = *(const u32x4*)p1;   // 8 bf16 (16B)
  return __builtin_bit_cast(v16bf, p);
}

__device__ __forceinline__ unsigned short f32_to_bf16(float f) {
  return (unsigned short)(__float_as_uint(f) >> 16);
}

// ---------------------------------------------------------------------------
// Tensor Data Mover: async-copy a [rows=128][cols=32] bf16 tile from global
// (row stride = K elements) into contiguous LDS. 2-D descriptor, 2 SGPR
// groups (VADDR2/3 = NULL), issued by one wave, tracked by TENSORcnt.
// Built per CDNA5 ISA.
// ---------------------------------------------------------------------------
__device__ __forceinline__ void tdm_load_tile_bf16(const unsigned short* gsrc,
                                                   unsigned int lds_byte_addr,
                                                   unsigned int k_stride_elems) {
  const unsigned long long ga = (unsigned long long)(uintptr_t)gsrc;
  u32x4 g0;
  g0[0] = 1u;                                        // count=1, user mode
  g0[1] = lds_byte_addr;                             // lds_addr
  g0[2] = (unsigned int)ga;                          // global_addr[31:0]
  g0[3] = (unsigned int)(ga >> 32) | (2u << 30);     // global_addr[56:32] | type=2
  u32x8 g1;
  g1[0] = 0x00010000u;        // workgroup_mask=0, data_size=1 (2 bytes)
  g1[1] = (32u << 16);        // tensor_dim0[15:0]=32 (bits 63:48)
  g1[2] = (128u << 16);       // tensor_dim0 hi=0, tensor_dim1[15:0]=128 (95:80)
  g1[3] = (32u << 16);        // tensor_dim1 hi=0, tile_dim0=32 (127:112)
  g1[4] = 128u;               // tile_dim1=128, tile_dim2=0
  g1[5] = k_stride_elems;     // tensor_dim0_stride[31:0] (data_size units)
  g1[6] = 0u;                 // stride hi, tensor_dim1_stride lo
  g1[7] = 0u;
  asm volatile("tensor_load_to_lds %0, %1" :: "s"(g0), "s"(g1) : "memory");
}

// ---------------------------------------------------------------------------
// Kernel 1: masked segment-mean of 3 embedding tables -> bf16 u[B][H]
// ---------------------------------------------------------------------------
__global__ __launch_bounds__(256)
void embed_mean_kernel(const float* __restrict__ itemE,
                       const float* __restrict__ entE,
                       const float* __restrict__ wordE,
                       const int* __restrict__ itemIds, const int* __restrict__ itemLen,
                       const int* __restrict__ entIds,  const int* __restrict__ entLen,
                       const int* __restrict__ wordIds, const int* __restrict__ wordLen,
                       unsigned short* __restrict__ u_out) {
  const int b  = blockIdx.x;
  const int li = itemLen[b], le = entLen[b], lw = wordLen[b];
  const float ri = 1.0f / (float)max(li, 1);
  const float re = 1.0f / (float)max(le, 1);
  const float rw = 1.0f / (float)max(lw, 1);
  for (int c = threadIdx.x; c < H_DIM; c += blockDim.x) {
    float si = 0.f, se = 0.f, sw = 0.f;
    for (int l = 0; l < li; ++l)
      si += itemE[(size_t)itemIds[b * SEQ_L + l] * H_DIM + c];
    for (int l = 0; l < le; ++l)
      se += entE[(size_t)entIds[b * SEQ_L + l] * H_DIM + c];
    for (int l = 0; l < lw; ++l)
      sw += wordE[(size_t)wordIds[b * SEQ_L + l] * H_DIM + c];
    const float u = (si * ri + se * re + sw * rw) * (1.0f / 3.0f);
    u_out[(size_t)b * H_DIM + c] = f32_to_bf16(u);
  }
}

// ---------------------------------------------------------------------------
// Kernel 2/3: bf16 WMMA GEMM.  A [M][K] bf16 (streamed to LDS by the TDM),
// Bw [K][N] f32 (converted to bf16 on the global->LDS path), C = A*B + bias.
// Workgroup tile 128x128, 8 waves (wave32) in a 4x2 grid, wave tile 32x64,
// K-step 32 -> v_wmma_f32_16x16x32_bf16, 8 WMMAs/wave/K-step.
// RELU_BF16_OUT=1: ReLU + bf16 store (hidden layer)
// RELU_BF16_OUT=0: f32 logits store
// ---------------------------------------------------------------------------
template <int RELU_BF16_OUT>
__global__ __launch_bounds__(256)
void gemm_bf16_wmma_kernel(const unsigned short* __restrict__ A,
                           const float* __restrict__ Bw,
                           const float* __restrict__ bias,
                           void* __restrict__ Cout,
                           int M, int N, int K) {
  __shared__ unsigned short ldsA[128][32];   // [m][k]  8 KB  (TDM destination)
  __shared__ unsigned short ldsBt[128][32];  // [n][k]  8 KB  (transposed)

  const int tileN = blockIdx.x * 128;
  const int tileM = blockIdx.y * 128;

  const int t    = threadIdx.x;
  const int lane = t & 31;
  const int wave = t >> 5;
  const int wm   = wave >> 1;       // 0..3 -> M offset wm*32
  const int wn   = wave & 1;        // 0..1 -> N offset wn*64
  const int half = lane >> 4;       // lane 0-15 vs 16-31
  const int lidx = lane & 15;
  const int k0   = half * 8;        // per-lane K chunk start (ISA layout)

  const unsigned int ldsA_addr =
      (unsigned int)(uintptr_t)&ldsA[0][0];          // generic addr low 32 = LDS byte addr
  const bool fullN = (tileN + 128 <= N);             // block-uniform

  v8f acc[2][4] = {};

  // B-tile loader: 32x128 f32 -> bf16 transposed, 16 floats per thread
  const int bk  = t >> 3;
  const int bn0 = (t & 7) * 16;

  for (int kk = 0; kk < K; kk += 32) {
    // --- A tile: one wave drives the Tensor Data Mover ---------------------
    if (wave == 0) {
      tdm_load_tile_bf16(A + (size_t)tileM * K + kk, ldsA_addr,
                         (unsigned int)K);
    }

    // --- B tile: all 256 threads load f32, convert, store transposed ------
    const float* bsrc = Bw + (size_t)(kk + bk) * N + tileN + bn0;
    if (fullN) {
#pragma unroll
      for (int q = 0; q < 4; ++q) {
        const float4 f = *(const float4*)(bsrc + q * 4);
        ldsBt[bn0 + q * 4 + 0][bk] = f32_to_bf16(f.x);
        ldsBt[bn0 + q * 4 + 1][bk] = f32_to_bf16(f.y);
        ldsBt[bn0 + q * 4 + 2][bk] = f32_to_bf16(f.z);
        ldsBt[bn0 + q * 4 + 3][bk] = f32_to_bf16(f.w);
      }
    } else {
      const float* brow = Bw + (size_t)(kk + bk) * N;
#pragma unroll
      for (int i = 0; i < 16; ++i) {
        const int gn = tileN + bn0 + i;
        const int cg = (gn < N) ? gn : (N - 1);      // clamp: load stays valid
        float v = brow[cg];                          // unconditional load
        v = (gn < N) ? v : 0.0f;                     // v_cndmask, no exec mask
        ldsBt[bn0 + i][bk] = f32_to_bf16(v);
      }
    }

    if (kk + 32 < K)                                 // prefetch next B tile
      __builtin_prefetch(bsrc + (size_t)32 * N, 0, 1);

    if (wave == 0) __builtin_amdgcn_s_wait_tensorcnt(0);  // TDM tile landed
    __syncthreads();

    // --- fragments + 8 WMMAs ----------------------------------------------
    v16bf afrag[2], bfrag[4];
#pragma unroll
    for (int i = 0; i < 2; ++i) {
      const unsigned short* r = &ldsA[wm * 32 + i * 16 + lidx][0];
      afrag[i] = make_frag(r + k0, r + 16 + k0);
    }
#pragma unroll
    for (int j = 0; j < 4; ++j) {
      const unsigned short* r = &ldsBt[wn * 64 + j * 16 + lidx][0];
      bfrag[j] = make_frag(r + k0, r + 16 + k0);
    }
#pragma unroll
    for (int i = 0; i < 2; ++i)
#pragma unroll
      for (int j = 0; j < 4; ++j)
        acc[i][j] = __builtin_amdgcn_wmma_f32_16x16x32_bf16(
            false, afrag[i], false, bfrag[j], (short)0, acc[i][j], false, false);
    __syncthreads();
  }

  // Epilogue. C/D layout: N = lane&15, M = vgpr + 8*(lane>=16).
#pragma unroll
  for (int i = 0; i < 2; ++i) {
#pragma unroll
    for (int j = 0; j < 4; ++j) {
      const int col = tileN + wn * 64 + j * 16 + lidx;
      if (col >= N) continue;
      const float bv = bias[col];
#pragma unroll
      for (int v = 0; v < 8; ++v) {
        const int row = tileM + wm * 32 + i * 16 + half * 8 + v;
        float val = acc[i][j][v] + bv;
        if (RELU_BF16_OUT) {
          val = fmaxf(val, 0.0f);
          ((unsigned short*)Cout)[(size_t)row * N + col] = f32_to_bf16(val);
        } else {
          ((float*)Cout)[(size_t)row * N + col] = val;
        }
      }
    }
  }
}

// ---------------------------------------------------------------------------
// Kernel 4: in-place row softmax over N columns (one block per row)
// ---------------------------------------------------------------------------
__global__ __launch_bounds__(256)
void softmax_kernel(float* __restrict__ logits, int N) {
  __shared__ float red[256];
  const int row = blockIdx.x;
  float* x = logits + (size_t)row * N;
  const int t = threadIdx.x;

  float m = -3.4e38f;
  for (int i = t; i < N; i += 256) m = fmaxf(m, x[i]);
  red[t] = m; __syncthreads();
  for (int s = 128; s; s >>= 1) { if (t < s) red[t] = fmaxf(red[t], red[t + s]); __syncthreads(); }
  m = red[0]; __syncthreads();

  float sum = 0.f;
  for (int i = t; i < N; i += 256) sum += __expf(x[i] - m);
  red[t] = sum; __syncthreads();
  for (int s = 128; s; s >>= 1) { if (t < s) red[t] += red[t + s]; __syncthreads(); }
  const float inv = 1.0f / red[0]; __syncthreads();

  for (int i = t; i < N; i += 256) x[i] = __expf(x[i] - m) * inv;
}

// ---------------------------------------------------------------------------
// Kernel 5: loss slot init (harness does not re-poison between replays)
// ---------------------------------------------------------------------------
__global__ void zero_loss_kernel(float* __restrict__ p) {
  if (threadIdx.x == 0) *p = 0.0f;
}

// ---------------------------------------------------------------------------
// Kernel 6: loss = -mean(log_softmax(probs)[label]); also emits labels-as-f32
// log_softmax(p) = p - logsumexp(p)
// ---------------------------------------------------------------------------
__global__ __launch_bounds__(256)
void loss_kernel(const float* __restrict__ probs,
                 const int* __restrict__ labels,
                 float* __restrict__ out_labels,
                 float* __restrict__ out_loss,
                 int N, float invB) {
  __shared__ float red[256];
  const int row = blockIdx.x;
  const float* x = probs + (size_t)row * N;
  const int t = threadIdx.x;

  float m = -3.4e38f;
  for (int i = t; i < N; i += 256) m = fmaxf(m, x[i]);
  red[t] = m; __syncthreads();
  for (int s = 128; s; s >>= 1) { if (t < s) red[t] = fmaxf(red[t], red[t + s]); __syncthreads(); }
  m = red[0]; __syncthreads();

  float sum = 0.f;
  for (int i = t; i < N; i += 256) sum += __expf(x[i] - m);
  red[t] = sum; __syncthreads();
  for (int s = 128; s; s >>= 1) { if (t < s) red[t] += red[t + s]; __syncthreads(); }

  if (t == 0) {
    const int lab = labels[row];
    const float lse  = m + __logf(red[0]);
    const float logp = x[lab] - lse;
    out_labels[row] = (float)lab;
    atomicAdd(out_loss, -logp * invB);
  }
}

// ---------------------------------------------------------------------------
extern "C" void kernel_launch(void* const* d_in, const int* in_sizes, int n_in,
                              void* d_out, int out_size, void* d_ws, size_t ws_size,
                              hipStream_t stream) {
  (void)in_sizes; (void)n_in; (void)out_size; (void)ws_size;

  // setup_inputs() dict order
  const int*   item_ids = (const int*)d_in[0];
  const int*   item_len = (const int*)d_in[1];
  const int*   ent_ids  = (const int*)d_in[2];
  const int*   ent_len  = (const int*)d_in[3];
  const int*   word_ids = (const int*)d_in[4];
  const int*   word_len = (const int*)d_in[5];
  const int*   labels   = (const int*)d_in[6];
  const float* item_emb = (const float*)d_in[7];
  const float* ent_emb  = (const float*)d_in[8];
  const float* word_emb = (const float*)d_in[9];
  const float* W1       = (const float*)d_in[10];
  const float* b1       = (const float*)d_in[11];
  const float* W2       = (const float*)d_in[12];
  const float* b2       = (const float*)d_in[13];

  // d_out layout: probs [B][NITEM] f32, then labels-as-f32 [B], then loss [1]
  float* probs      = (float*)d_out;
  float* out_labels = probs + (size_t)B_SZ * NITEM;
  float* out_loss   = out_labels + B_SZ;

  // workspace: u bf16 [B][H], h bf16 [B][4H]  (5 MB total)
  unsigned short* u = (unsigned short*)d_ws;
  unsigned short* h = u + (size_t)B_SZ * H_DIM;

  // 1) embedding masked-mean -> bf16 u
  embed_mean_kernel<<<B_SZ, 256, 0, stream>>>(
      item_emb, ent_emb, word_emb,
      item_ids, item_len, ent_ids, ent_len, word_ids, word_len, u);

  // 2) h = relu(u @ W1 + b1) -> bf16  (M=1024, N=2048, K=512)
  gemm_bf16_wmma_kernel<1><<<dim3(H4 / 128, B_SZ / 128), 256, 0, stream>>>(
      u, W1, b1, (void*)h, B_SZ, H4, H_DIM);

  // 3) logits = h @ W2 + b2 -> f32 into probs region (M=1024, N=50000, K=2048)
  gemm_bf16_wmma_kernel<0><<<dim3((NITEM + 127) / 128, B_SZ / 128), 256, 0, stream>>>(
      h, W2, b2, (void*)probs, B_SZ, NITEM, H4);

  // 4) probs = softmax(logits), in place
  softmax_kernel<<<B_SZ, 256, 0, stream>>>(probs, NITEM);

  // 5) loss = -mean(log_softmax(probs)[label]); labels as f32
  zero_loss_kernel<<<1, 64, 0, stream>>>(out_loss);
  loss_kernel<<<B_SZ, 256, 0, stream>>>(probs, labels, out_labels, out_loss,
                                        NITEM, 1.0f / (float)B_SZ);
}
